// MultiHeadedAttentionFixedSpan_75436805587623
// MI455X (gfx1250) — matrix operand order repaired and checked by reading, not compiled, not run on hardware
//
#include <hip/hip_runtime.h>

// ---------------------------------------------------------------------------
// Types for CDNA5 WMMA (wave32, 16x16x32 bf16 -> f32)
// ---------------------------------------------------------------------------
typedef __bf16 bf16;
typedef __attribute__((ext_vector_type(16))) __bf16 v16bf;
typedef __attribute__((ext_vector_type(8)))  __bf16 v8bf;
typedef __attribute__((ext_vector_type(4)))  __bf16 v4bf;
typedef __attribute__((ext_vector_type(8)))  float  v8f;

union V16U  { v16bf v; v8bf h[2]; };

__device__ __forceinline__ v8f wmma_bf16(v16bf a, v16bf b, v8f c) {
    // (neg_a, A, neg_b, B, c_mod, C, reuse_a, reuse_b)
    return __builtin_amdgcn_wmma_f32_16x16x32_bf16(false, a, false, b,
                                                   (short)0, c, false, false);
}

// CDNA5 async global->LDS copy (16B per lane, tracked by ASYNCcnt).
__device__ __forceinline__ void async_copy_b128(const bf16* gsrc, bf16* ldst) {
    unsigned loff = (unsigned)(uintptr_t)ldst;   // low 32 bits of generic ptr = LDS offset
    asm volatile("global_load_async_to_lds_b128 %0, %1, off"
                 :: "v"(loff), "v"(gsrc)
                 : "memory");
}
__device__ __forceinline__ void wait_async() {
    asm volatile("s_wait_asynccnt 0x0" ::: "memory");
}

#define T_LEN 1024
#define F_LEN 512
#define NHEAD 8
#define DK    64
#define SPAN_ 100
#define PAD_L 50
#define VT_STRIDE 184   // halves; 92 dwords -> 16-lane fragment reads hit 16 banks

// ---------------------------------------------------------------------------
// Kernel 1: W (f32, [K][N] row-major) -> Wt (bf16, [N][K]) for q,k,v,o
// ---------------------------------------------------------------------------
__global__ __launch_bounds__(256) void k_transpose_weights(
    const float* __restrict__ Wq, const float* __restrict__ Wk,
    const float* __restrict__ Wv, const float* __restrict__ Wo,
    bf16* __restrict__ Wt)
{
    const float* W = (blockIdx.z == 0) ? Wq :
                     (blockIdx.z == 1) ? Wk :
                     (blockIdx.z == 2) ? Wv : Wo;
    bf16* out = Wt + (size_t)blockIdx.z * (F_LEN * F_LEN);
    int idx = blockIdx.x * 256 + threadIdx.x;       // grid.x = 512*512/256
    int n = idx >> 9;
    int k = idx & 511;
    out[(size_t)n * F_LEN + k] = (bf16)W[(size_t)k * F_LEN + n];
}

// ---------------------------------------------------------------------------
// Kernel 2: fused Q/K/V projection GEMM.
//   C(4096x512) = X(4096x512,f32) * W(512x512) + b  ->  bf16 [B][H][T][DK]
//   Block: 256 threads (8 waves), tile 128(M) x 64(N), K-step 32.
//   W^T tile staged with async global->LDS DMA; A tile converts f32->bf16.
// ---------------------------------------------------------------------------
__global__ __launch_bounds__(256) void k_proj_qkv(
    const float* __restrict__ Xq, const float* __restrict__ Xk,
    const float* __restrict__ Xv,
    const bf16* __restrict__ Wtq, const bf16* __restrict__ Wtk,
    const bf16* __restrict__ Wtv,
    const float* __restrict__ bq, const float* __restrict__ bk,
    const float* __restrict__ bv,
    bf16* __restrict__ Qo, bf16* __restrict__ Ko, bf16* __restrict__ Vo)
{
    const float* X; const bf16* Wt; const float* bias; bf16* Out;
    if (blockIdx.z == 0)      { X = Xq; Wt = Wtq; bias = bq; Out = Qo; }
    else if (blockIdx.z == 1) { X = Xk; Wt = Wtk; bias = bk; Out = Ko; }
    else                      { X = Xv; Wt = Wtv; bias = bv; Out = Vo; }

    __shared__ __align__(16) bf16 As[128 * 32];   // 8 KB, [m][k]
    __shared__ __align__(16) bf16 Bs[64 * 32];    // 4 KB, [n][k] (W^T tile)

    const int tid  = threadIdx.x;
    const int lane = tid & 31;
    const int wid  = tid >> 5;
    const int mr   = lane & 15;
    const int hh   = lane >> 4;        // half-wave id
    const int wm   = wid & 3;          // wave M group (4 x 32 = 128)
    const int wn   = wid >> 2;         // wave N group (2 x 32 = 64)
    const int m_blk = blockIdx.y * 128;
    const int n_blk = blockIdx.x * 64;

    const v8f vzero = {0.f,0.f,0.f,0.f,0.f,0.f,0.f,0.f};
    v8f acc[2][2];
    acc[0][0] = vzero; acc[0][1] = vzero; acc[1][0] = vzero; acc[1][1] = vzero;

    for (int kt = 0; kt < F_LEN; kt += 32) {
        __syncthreads();
        // B tile: 64x32 bf16 via async global->LDS (1 instr / lane)
        {
            int row = tid >> 2;
            int seg = (tid & 3) * 8;
            async_copy_b128(Wt + (size_t)(n_blk + row) * F_LEN + kt + seg,
                            Bs + row * 32 + seg);
        }
        // A tile: 128x32 f32 -> bf16 (1024 float4 / 256 threads)
        #pragma unroll
        for (int i = 0; i < 4; ++i) {
            int id  = tid + i * 256;
            int row = id >> 3;
            int col = (id & 7) * 4;
            const float4 f = *(const float4*)(X + (size_t)(m_blk + row) * F_LEN + kt + col);
            v4bf c4 = { (bf16)f.x, (bf16)f.y, (bf16)f.z, (bf16)f.w };
            *(v4bf*)(As + row * 32 + col) = c4;
        }
        // prefetch next-iteration A rows into cache
        if (kt + 32 < F_LEN)
            __builtin_prefetch(X + (size_t)(m_blk + (tid & 127)) * F_LEN + kt + 32, 0, 1);
        wait_async();
        __syncthreads();

        v16bf afrag[2], bfrag[2];
        #pragma unroll
        for (int i = 0; i < 2; ++i) {
            const bf16* ar = As + (wm * 32 + i * 16 + mr) * 32;
            V16U ua;
            ua.h[0] = *(const v8bf*)(ar + hh * 8);        // K = hi*8 + (0..7)
            ua.h[1] = *(const v8bf*)(ar + 16 + hh * 8);   // K = 16 + hi*8 + (0..7)
            afrag[i] = ua.v;
            const bf16* br = Bs + (wn * 32 + i * 16 + mr) * 32;
            V16U ub;
            ub.h[0] = *(const v8bf*)(br + hh * 8);        // K = hi*8 + (0..7)
            ub.h[1] = *(const v8bf*)(br + 16 + hh * 8);   // K = 16 + hi*8 + (0..7)
            bfrag[i] = ub.v;
        }
        #pragma unroll
        for (int i = 0; i < 2; ++i)
            #pragma unroll
            for (int j = 0; j < 2; ++j)
                acc[i][j] = wmma_bf16(afrag[i], bfrag[j], acc[i][j]);
    }

    // Store bf16 into [B][H][T][DK]
    #pragma unroll
    for (int i = 0; i < 2; ++i) {
        #pragma unroll
        for (int j = 0; j < 2; ++j) {
            int nbase = n_blk + wn * 32 + j * 16 + mr;
            float bval = bias[nbase];
            int hd = nbase >> 6;
            int d  = nbase & 63;
            #pragma unroll
            for (int r = 0; r < 8; ++r) {
                int m = m_blk + wm * 32 + i * 16 + hh * 8 + r;
                int bb = m >> 10;
                int t  = m & 1023;
                Out[(((size_t)bb * NHEAD + hd) * T_LEN + t) * DK + d] =
                    (bf16)(acc[i][j][r] + bval);
            }
        }
    }
}

// ---------------------------------------------------------------------------
// Kernel 3: banded attention. Grid (T/64, H, B), block = 128 (4 waves).
// Wave w handles query rows t0..t0+15, window = 128 key positions
// [t0-50, t0+77]; valid span per row i is cols [i, i+99] ∩ [0,T).
// V window staged TRANSPOSED (Vt[dk][row]) so PV B-fragments are b128 reads.
// ---------------------------------------------------------------------------
__global__ __launch_bounds__(128) void k_attn(
    const bf16* __restrict__ Q, const bf16* __restrict__ K,
    const bf16* __restrict__ V, const int* __restrict__ mask,
    bf16* __restrict__ Xout)
{
    __shared__ __align__(16) bf16 Ps[4][16 * 128];      // 16 KB: per-wave P
    __shared__ __align__(16) bf16 Vt[DK * VT_STRIDE];   // ~23 KB: V^T window

    const int tid  = threadIdx.x;
    const int lane = tid & 31;
    const int w    = tid >> 5;
    const int mr   = lane & 15;
    const int hh   = lane >> 4;
    const int hd   = blockIdx.y;
    const int bb   = blockIdx.z;
    const int tb   = blockIdx.x * 64;
    const int t0   = tb + w * 16;

    const size_t headoff = ((size_t)bb * NHEAD + hd) * T_LEN * DK;
    const bf16* Qb = Q + headoff;
    const bf16* Kb = K + headoff;
    const bf16* Vb = V + headoff;

    // Stage block V window transposed: rows tb-50 .. tb+125 (176 rows,
    // clamped; out-of-band rows get P==0 so their values are harmless).
    for (int id = tid; id < 176 * 8; id += 128) {
        int row  = id >> 3;
        int dseg = (id & 7) * 8;
        int p    = tb - PAD_L + row;
        int pc   = min(max(p, 0), T_LEN - 1);
        v8bf vv = *(const v8bf*)(Vb + (size_t)pc * DK + dseg);
        #pragma unroll
        for (int j = 0; j < 8; ++j)
            Vt[(dseg + j) * VT_STRIDE + row] = vv[j];
    }

    // ---- scores: S(16x128) = Q_tile(16x64) * K_window(128x64)^T ----
    const v8f vzero = {0.f,0.f,0.f,0.f,0.f,0.f,0.f,0.f};
    v8f S[8];
    #pragma unroll
    for (int jt = 0; jt < 8; ++jt) S[jt] = vzero;

    #pragma unroll
    for (int kk = 0; kk < DK; kk += 32) {
        const bf16* qr = Qb + (size_t)(t0 + mr) * DK + kk;
        V16U ua;
        ua.h[0] = *(const v8bf*)(qr + hh * 16);
        ua.h[1] = *(const v8bf*)(qr + hh * 16 + 8);
        #pragma unroll
        for (int jt = 0; jt < 8; ++jt) {
            int p  = t0 - PAD_L + jt * 16 + mr;
            int pc = min(max(p, 0), T_LEN - 1);
            const bf16* kr = Kb + (size_t)pc * DK + kk + hh * 16;
            V16U ub;
            ub.h[0] = *(const v8bf*)kr;
            ub.h[1] = *(const v8bf*)(kr + 8);
            S[jt] = wmma_bf16(ua.v, ub.v, S[jt]);
        }
    }

    // Key-position validity per window column group (independent of row r)
    bool kvalid[8];
    #pragma unroll
    for (int jt = 0; jt < 8; ++jt) {
        int p = t0 - PAD_L + jt * 16 + mr;
        int pc = min(max(p, 0), T_LEN - 1);
        kvalid[jt] = (p >= 0) && (p < T_LEN) && (mask[bb * T_LEN + pc] != 0);
    }

    // ---- masked row softmax (row m = hh*8+r lives in one 16-lane half) ----
    const float scale = 0.125f;   // 1/sqrt(64)
    #pragma unroll
    for (int r = 0; r < 8; ++r) {
        int i = hh * 8 + r;
        float mx = -3.0e38f;
        float sv[8];
        #pragma unroll
        for (int jt = 0; jt < 8; ++jt) {
            int col = jt * 16 + mr;
            bool valid = kvalid[jt] && (col >= i) && (col <= i + SPAN_ - 1);
            float s = valid ? S[jt][r] * scale : -3.0e38f;
            sv[jt] = s;
            mx = fmaxf(mx, s);
        }
        #pragma unroll
        for (int off = 1; off < 16; off <<= 1)
            mx = fmaxf(mx, __shfl_xor(mx, off, 32));
        float sum = 0.f;
        #pragma unroll
        for (int jt = 0; jt < 8; ++jt) {
            float e = (sv[jt] > -1.0e38f) ? __expf(sv[jt] - mx) : 0.f;
            sv[jt] = e;
            sum += e;
        }
        #pragma unroll
        for (int off = 1; off < 16; off <<= 1)
            sum += __shfl_xor(sum, off, 32);
        float inv = 1.0f / sum;
        #pragma unroll
        for (int jt = 0; jt < 8; ++jt)
            Ps[w][i * 128 + jt * 16 + mr] = (bf16)(sv[jt] * inv);
    }
    __syncthreads();   // Ps ready (all waves) + Vt ready (cooperative)

    // ---- O(16x64) = P(16x128) * V_window(128x64) ----
    v16bf pa[4];
    #pragma unroll
    for (int kk4 = 0; kk4 < 4; ++kk4) {
        const bf16* pr = &Ps[w][mr * 128 + kk4 * 32];
        V16U u;
        u.h[0] = *(const v8bf*)(pr + hh * 16);
        u.h[1] = *(const v8bf*)(pr + hh * 16 + 8);
        pa[kk4] = u.v;
    }
    v8f O[4];
    #pragma unroll
    for (int nt = 0; nt < 4; ++nt) O[nt] = vzero;
    #pragma unroll
    for (int nt = 0; nt < 4; ++nt) {
        int n = nt * 16 + mr;                 // dk column for this lane
        #pragma unroll
        for (int kk4 = 0; kk4 < 4; ++kk4) {
            // window rows w*16 + kk4*32 + hh*16 + (0..15), contiguous in Vt
            const bf16* vr = Vt + n * VT_STRIDE + w * 16 + kk4 * 32 + hh * 16;
            V16U ub;
            ub.h[0] = *(const v8bf*)vr;
            ub.h[1] = *(const v8bf*)(vr + 8);
            O[nt] = wmma_bf16(pa[kk4], ub.v, O[nt]);
        }
    }

    // Store bf16 x (row-major 4096 x 512), column = head*64 + dk
    #pragma unroll
    for (int nt = 0; nt < 4; ++nt) {
        int n = hd * DK + nt * 16 + mr;
        #pragma unroll
        for (int r = 0; r < 8; ++r) {
            int t = t0 + hh * 8 + r;
            Xout[((size_t)(bb * T_LEN + t)) * F_LEN + n] = (bf16)O[nt][r];
        }
    }
}

// ---------------------------------------------------------------------------
// Kernel 4: output projection. out(4096x512,f32) = Xbf * Wo + bo
// Both tiles are bf16 -> both staged with async global->LDS DMA.
// ---------------------------------------------------------------------------
__global__ __launch_bounds__(256) void k_out_proj(
    const bf16* __restrict__ Xbf, const bf16* __restrict__ Wto,
    const float* __restrict__ bo, float* __restrict__ Out)
{
    __shared__ __align__(16) bf16 As[128 * 32];
    __shared__ __align__(16) bf16 Bs[64 * 32];

    const int tid  = threadIdx.x;
    const int lane = tid & 31;
    const int wid  = tid >> 5;
    const int mr   = lane & 15;
    const int hh   = lane >> 4;
    const int wm   = wid & 3;
    const int wn   = wid >> 2;
    const int m_blk = blockIdx.y * 128;
    const int n_blk = blockIdx.x * 64;

    const v8f vzero = {0.f,0.f,0.f,0.f,0.f,0.f,0.f,0.f};
    v8f acc[2][2];
    acc[0][0] = vzero; acc[0][1] = vzero; acc[1][0] = vzero; acc[1][1] = vzero;

    for (int kt = 0; kt < F_LEN; kt += 32) {
        __syncthreads();
        #pragma unroll
        for (int i = 0; i < 2; ++i) {          // A tile: 128x32 bf16 (512 x b128)
            int id  = tid + i * 256;
            int row = id >> 2;
            int seg = (id & 3) * 8;
            async_copy_b128(Xbf + (size_t)(m_blk + row) * F_LEN + kt + seg,
                            As + row * 32 + seg);
        }
        {                                      // B tile: 64x32 bf16 (256 x b128)
            int row = tid >> 2;
            int seg = (tid & 3) * 8;
            async_copy_b128(Wto + (size_t)(n_blk + row) * F_LEN + kt + seg,
                            Bs + row * 32 + seg);
        }
        wait_async();
        __syncthreads();

        v16bf afrag[2], bfrag[2];
        #pragma unroll
        for (int i = 0; i < 2; ++i) {
            const bf16* ar = As + (wm * 32 + i * 16 + mr) * 32;
            V16U ua;
            ua.h[0] = *(const v8bf*)(ar + hh * 8);
            ua.h[1] = *(const v8bf*)(ar + 16 + hh * 8);
            afrag[i] = ua.v;
            const bf16* br = Bs + (wn * 32 + i * 16 + mr) * 32;
            V16U ub;
            ub.h[0] = *(const v8bf*)(br + hh * 8);
            ub.h[1] = *(const v8bf*)(br + 16 + hh * 8);
            bfrag[i] = ub.v;
        }
        #pragma unroll
        for (int i = 0; i < 2; ++i)
            #pragma unroll
            for (int j = 0; j < 2; ++j)
                acc[i][j] = wmma_bf16(afrag[i], bfrag[j], acc[i][j]);
    }

    #pragma unroll
    for (int i = 0; i < 2; ++i) {
        #pragma unroll
        for (int j = 0; j < 2; ++j) {
            int nbase = n_blk + wn * 32 + j * 16 + mr;
            float bval = bo[nbase];
            #pragma unroll
            for (int r = 0; r < 8; ++r) {
                int m = m_blk + wm * 32 + i * 16 + hh * 8 + r;
                Out[(size_t)m * F_LEN + nbase] = acc[i][j][r] + bval;
            }
        }
    }
}

// ---------------------------------------------------------------------------
// Launch
// ---------------------------------------------------------------------------
extern "C" void kernel_launch(void* const* d_in, const int* in_sizes, int n_in,
                              void* d_out, int out_size, void* d_ws, size_t ws_size,
                              hipStream_t stream) {
    (void)in_sizes; (void)n_in; (void)out_size; (void)ws_size;
    const float* query = (const float*)d_in[0];
    const float* key   = (const float*)d_in[1];
    const float* value = (const float*)d_in[2];
    const int*   mask  = (const int*)d_in[3];
    const float* Wq = (const float*)d_in[4];
    const float* bq = (const float*)d_in[5];
    const float* Wk = (const float*)d_in[6];
    const float* bk = (const float*)d_in[7];
    const float* Wv = (const float*)d_in[8];
    const float* bv = (const float*)d_in[9];
    const float* Wo = (const float*)d_in[10];
    const float* bo = (const float*)d_in[11];
    float* out = (float*)d_out;

    char* ws = (char*)d_ws;
    bf16* wt  = (bf16*)(ws);                                  // 4 x 512x512 bf16 (2 MB)
    bf16* Qb  = (bf16*)(ws + 2097152);                        // 4 MB
    bf16* Kb  = (bf16*)(ws + 2097152 + 4194304);              // 4 MB
    bf16* Vb  = (bf16*)(ws + 2097152 + 2 * 4194304);          // 4 MB
    bf16* Xbf = (bf16*)(ws + 2097152 + 3 * 4194304);          // 4 MB

    k_transpose_weights<<<dim3(1024, 1, 4), 256, 0, stream>>>(Wq, Wk, Wv, Wo, wt);
    k_proj_qkv<<<dim3(8, 32, 3), 256, 0, stream>>>(
        query, key, value,
        wt, wt + 262144, wt + 2 * 262144,
        bq, bk, bv, Qb, Kb, Vb);
    k_attn<<<dim3(16, 8, 4), 128, 0, stream>>>(Qb, Kb, Vb, mask, Xbf);
    k_out_proj<<<dim3(8, 32), 256, 0, stream>>>(Xbf, wt + 3 * 262144, bo, out);
}
